// Mask2FormerPixelDecoder_71330816852924
// MI455X (gfx1250) — matrix operand order repaired and checked by reading, not compiled, not run on hardware
//
#include <hip/hip_runtime.h>

#define SEQn 5376
#define Bn   4
#define Dn   256
#define NHn  8
#define NLn  3
#define NPn  4
#define DHn  32

typedef __attribute__((ext_vector_type(16))) __bf16 v16bf;
typedef __attribute__((ext_vector_type(8)))  float  v8f;
typedef __attribute__((ext_vector_type(4)))  unsigned int v4u;
typedef __attribute__((ext_vector_type(8)))  unsigned int v8u;

union FragBF { v16bf v; unsigned int u[8]; };

__device__ __forceinline__ unsigned short f2bf(float f) {
    unsigned int u = __float_as_uint(f);
    u += 0x7fffu + ((u >> 16) & 1u);   // round-to-nearest-even
    return (unsigned short)(u >> 16);
}

__device__ __forceinline__ int imin(int a, int b) { return a < b ? a : b; }
__device__ __forceinline__ int imax(int a, int b) { return a > b ? a : b; }

// ---------------------------------------------------------------- prep / cvt
__global__ void prep_kernel(const float* __restrict__ hid, const float* __restrict__ pos,
                            unsigned short* __restrict__ hid_bf,
                            unsigned short* __restrict__ hs_bf, int n) {
    int i = blockIdx.x * blockDim.x + threadIdx.x;
    if (i < n) {
        float h = hid[i];
        hid_bf[i] = f2bf(h);
        hs_bf[i]  = f2bf(h + pos[i]);
    }
}

// W[K][N] f32  ->  Wt[N][K] bf16 (transposed so GEMM B tiles are row-major in K)
__global__ void cvt_wt_kernel(const float* __restrict__ W, unsigned short* __restrict__ Wt,
                              int K, int N) {
    int i = blockIdx.x * blockDim.x + threadIdx.x;
    if (i < K * N) {
        int k = i / N, n = i % N;
        Wt[(size_t)n * K + k] = f2bf(W[i]);
    }
}

// ---------------------------------------------------------------- TDM + WMMA GEMM
// C[M,N] = A[M,K] * Wt[N,K]^T + bias[N];  A,Wt bf16 row-major; C f32.
// grid = (N/32, M/128), block = 256 threads = 8 waves (wave32).
// Block tile 128x32, K-step 32. Wave w owns rows [w*16,w*16+16), all 32 cols
// -> 2 WMMAs per wave per K-step sharing one A fragment.
// Tiles are DMA'd global->LDS by the Tensor Data Mover, DOUBLE-BUFFERED:
// wave 0 waits tensorcnt for tile i, one barrier publishes it, then issues tile
// i+1 into the other buffer while all waves compute on tile i. Descriptor pad
// fields insert 8B per 64B row -> LDS row stride 36 elems (conflict-free reads).
__global__ __launch_bounds__(256) void gemm_bf16_tdm_kernel(
    const unsigned short* __restrict__ A,   // [M][K] bf16
    const unsigned short* __restrict__ Wt,  // [N][K] bf16
    const float* __restrict__ bias, float* __restrict__ C, int M, int N, int K)
{
    const int AS = 36;                           // padded LDS row stride (elems)
    __shared__ unsigned short At[2][128 * 36];   // A tile [m][k], double-buffered
    __shared__ unsigned short Bt[2][32 * 36];    // B tile [n][k], double-buffered

    const int tid   = threadIdx.x;
    const int wave  = tid >> 5;
    const int lane  = tid & 31;
    const int khalf = lane >> 4;                 // 0/1
    const int l16   = lane & 15;

    const int m0 = blockIdx.y * 128;
    const int n0 = blockIdx.x * 32;

    v8f acc0 = {}, acc1 = {};

    // D# group1 (loop-invariant): data_size=2B, pad_enable, pad_interval=16 dwords
    // (code 3), pad_amount=2 dwords (code 1)  -> +8B after each 64B tile row.
    const unsigned int cfg = (1u << 16) | (1u << 20) | (3u << 22) | (1u << 25);
    v8u d1a, d1b;
    d1a[0] = cfg;
    d1a[1] = ((unsigned int)K & 0xffffu) << 16;                        // tensor_dim0 lo
    d1a[2] = ((unsigned int)K >> 16) | (((unsigned int)M & 0xffffu) << 16);
    d1a[3] = ((unsigned int)M >> 16) | (32u << 16);                    // tile_dim0 = 32
    d1a[4] = 128u;                                                     // tile_dim1 = 128
    d1a[5] = (unsigned int)K;                                          // dim0 stride
    d1a[6] = 0u; d1a[7] = 0u;
    d1b = d1a;
    d1b[2] = ((unsigned int)K >> 16) | (((unsigned int)N & 0xffffu) << 16);
    d1b[3] = ((unsigned int)N >> 16) | (32u << 16);
    d1b[4] = 32u;                                                      // tile_dim1 = 32

    const unsigned short* Abase = A  + (size_t)m0 * K;
    const unsigned short* Bbase = Wt + (size_t)n0 * K;
    const unsigned int ldsA[2] = { (unsigned int)(size_t)At[0], (unsigned int)(size_t)At[1] };
    const unsigned int ldsB[2] = { (unsigned int)(size_t)Bt[0], (unsigned int)(size_t)Bt[1] };

    auto issue_tile = [&](int k0, int buf) {
        unsigned long long gaA = (unsigned long long)(size_t)(Abase + k0);
        unsigned long long gaB = (unsigned long long)(size_t)(Bbase + k0);
        v4u d0a, d0b;
        // group0: count=1 | lds_addr | global_addr[56:0] | type=2
        d0a[0] = 1u; d0a[1] = ldsA[buf];
        d0a[2] = (unsigned int)gaA;
        d0a[3] = ((unsigned int)(gaA >> 32) & 0x01ffffffu) | (2u << 30);
        d0b[0] = 1u; d0b[1] = ldsB[buf];
        d0b[2] = (unsigned int)gaB;
        d0b[3] = ((unsigned int)(gaB >> 32) & 0x01ffffffu) | (2u << 30);
        asm volatile("tensor_load_to_lds %0, %1" :: "s"(d0a), "s"(d1a) : "memory");
        asm volatile("tensor_load_to_lds %0, %1" :: "s"(d0b), "s"(d1b) : "memory");
    };

    const int iters = K >> 5;
    if (wave == 0) issue_tile(0, 0);             // prologue: prefetch tile 0

    for (int i = 0; i < iters; ++i) {
        const int buf = i & 1;
        if (wave == 0) __builtin_amdgcn_s_wait_tensorcnt(0);   // tile i landed
        __syncthreads();                          // publish tile i; buf^1 reads done
        if ((wave == 0) && (i + 1 < iters)) issue_tile((i + 1) << 5, buf ^ 1);

        // A fragment: lanes 0-15 K 0..15, lanes 16-31 K 16..31 (matches B)
        FragBF a, b0, b1;
        const unsigned int* ar = (const unsigned int*)(At[buf] + (wave * 16 + l16) * AS);
        #pragma unroll
        for (int r = 0; r < 4; ++r) {
            a.u[r]     = ar[8 * khalf + r];
            a.u[4 + r] = ar[8 * khalf + 4 + r];
        }
        // B fragments from [n][k] tile: lanes 0-15 K 0..15, lanes 16-31 K 16..31
        const unsigned int* br0 = (const unsigned int*)(Bt[buf] + l16 * AS);
        const unsigned int* br1 = (const unsigned int*)(Bt[buf] + (16 + l16) * AS);
        #pragma unroll
        for (int r = 0; r < 8; ++r) { b0.u[r] = br0[8 * khalf + r]; b1.u[r] = br1[8 * khalf + r]; }

        acc0 = __builtin_amdgcn_wmma_f32_16x16x32_bf16(false, a.v, false, b0.v,
                                                       (short)0, acc0, false, false);
        acc1 = __builtin_amdgcn_wmma_f32_16x16x32_bf16(false, a.v, false, b1.v,
                                                       (short)0, acc1, false, false);
    }

    // C/D layout: VGPR r holds row (8*khalf + r), col = lane&15
    const int col0 = n0 + l16;
    const int col1 = n0 + 16 + l16;
    const float bv0 = bias[col0];
    const float bv1 = bias[col1];
    #pragma unroll
    for (int r = 0; r < 8; ++r) {
        int row = m0 + wave * 16 + khalf * 8 + r;
        C[(size_t)row * N + col0] = acc0[r] + bv0;
        C[(size_t)row * N + col1] = acc1[r] + bv1;
    }
}

// ---------------------------------------------------------------- softmax over 12
__global__ void softmax_kernel(const float* __restrict__ logits, float* __restrict__ attn, int n) {
    int i = blockIdx.x * blockDim.x + threadIdx.x;
    if (i >= n) return;
    const float* p = logits + (size_t)i * 12;
    float m = p[0];
    #pragma unroll
    for (int j = 1; j < 12; ++j) m = fmaxf(m, p[j]);
    float e[12], s = 0.f;
    #pragma unroll
    for (int j = 0; j < 12; ++j) { e[j] = __expf(p[j] - m); s += e[j]; }
    float inv = 1.f / s;
    float* o = attn + (size_t)i * 12;
    #pragma unroll
    for (int j = 0; j < 12; ++j) o[j] = e[j] * inv;
}

// ---------------------------------------------------------------- deformable sampling
// One wave per (b,q,h); lane = channel (DH==32). Coalesced b32 gathers per corner.
__global__ __launch_bounds__(256) void sample_kernel(
    const float* __restrict__ value,   // (B*SEQ, 256) f32
    const float* __restrict__ offv,    // (B*SEQ, 192) f32
    const float* __restrict__ attn,    // (B*SEQ*NH, 12) f32
    const float* __restrict__ ref,     // (B, SEQ, NL, 2) f32
    unsigned short* __restrict__ outp) // (B*SEQ, 256) bf16
{
    const int Wl[3]    = { 64, 32, 16 };   // square levels
    const int start[3] = { 0, 4096, 5120 };

    int wave  = threadIdx.x >> 5;
    int lane  = threadIdx.x & 31;
    int tuple = blockIdx.x * 8 + wave;     // (b*SEQ+q)*NH + h
    if (tuple >= Bn * SEQn * NHn) return;

    int bq = tuple >> 3;                   // b*SEQ + q
    int h  = tuple & 7;
    int b  = bq / SEQn;

    const float* rp = ref  + (size_t)bq * (NLn * 2);
    const float* op = offv + (size_t)bq * 192 + h * 24;
    const float* ap = attn + (size_t)tuple * 12;

    float acc = 0.f;
    #pragma unroll
    for (int l = 0; l < 3; ++l) {
        const int   Wi = Wl[l];
        const float fW = (float)Wi;
        const float rx = rp[l * 2 + 0];
        const float ry = rp[l * 2 + 1];
        const size_t vbase = ((size_t)(b * SEQn + start[l])) * 256 + h * 32 + lane;
        #pragma unroll
        for (int p = 0; p < 4; ++p) {
            float ox = op[l * 8 + p * 2 + 0];
            float oy = op[l * 8 + p * 2 + 1];
            float x = rx * fW + ox - 0.5f;   // loc*W - 0.5
            float y = ry * fW + oy - 0.5f;
            float x0f = floorf(x), y0f = floorf(y);
            int   x0 = (int)x0f,  y0 = (int)y0f;
            float wx = x - x0f,   wy = y - y0f;
            float aw = ap[l * 4 + p];

            bool xv0 = (x0 >= 0) && (x0 < Wi);
            bool xv1 = (x0 + 1 >= 0) && (x0 + 1 < Wi);
            bool yv0 = (y0 >= 0) && (y0 < Wi);
            bool yv1 = (y0 + 1 >= 0) && (y0 + 1 < Wi);
            int xc0 = imin(imax(x0, 0), Wi - 1);
            int xc1 = imin(imax(x0 + 1, 0), Wi - 1);
            int yc0 = imin(imax(y0, 0), Wi - 1);
            int yc1 = imin(imax(y0 + 1, 0), Wi - 1);

            float v00 = (xv0 && yv0) ? value[vbase + (size_t)(yc0 * Wi + xc0) * 256] : 0.f;
            float v01 = (xv1 && yv0) ? value[vbase + (size_t)(yc0 * Wi + xc1) * 256] : 0.f;
            float v10 = (xv0 && yv1) ? value[vbase + (size_t)(yc1 * Wi + xc0) * 256] : 0.f;
            float v11 = (xv1 && yv1) ? value[vbase + (size_t)(yc1 * Wi + xc1) * 256] : 0.f;

            acc += aw * ((1.f - wx) * (1.f - wy) * v00 + wx * (1.f - wy) * v01 +
                         (1.f - wx) * wy * v10 + wx * wy * v11);
        }
    }
    outp[(size_t)bq * 256 + h * 32 + lane] = f2bf(acc);
}

// ---------------------------------------------------------------- launcher
extern "C" void kernel_launch(void* const* d_in, const int* in_sizes, int n_in,
                              void* d_out, int out_size, void* d_ws, size_t ws_size,
                              hipStream_t stream) {
    const float* hidden = (const float*)d_in[0];
    const float* pos    = (const float*)d_in[1];
    const float* ref    = (const float*)d_in[2];
    const float* Wv     = (const float*)d_in[3];
    const float* bv     = (const float*)d_in[4];
    const float* Wo     = (const float*)d_in[5];
    const float* bo     = (const float*)d_in[6];
    const float* Wa     = (const float*)d_in[7];
    const float* ba     = (const float*)d_in[8];
    const float* Wout   = (const float*)d_in[9];
    const float* bout   = (const float*)d_in[10];

    float* out      = (float*)d_out;
    float* attn_out = out + (size_t)Bn * SEQn * Dn;   // second tuple output

    const int MT = Bn * SEQn;                         // 21504 (multiple of 128)

    char*  ws  = (char*)d_ws;
    size_t pos_ = 0;
    auto alloc = [&](size_t bytes) -> void* {
        void* p = ws + pos_;
        pos_ += (bytes + 255) & ~(size_t)255;
        return p;
    };
    unsigned short* hid_bf  = (unsigned short*)alloc((size_t)MT * Dn * 2);
    unsigned short* hs_bf   = (unsigned short*)alloc((size_t)MT * Dn * 2);
    unsigned short* WvT     = (unsigned short*)alloc((size_t)Dn * Dn * 2);   // [N][K]
    unsigned short* WoT     = (unsigned short*)alloc((size_t)192 * Dn * 2);
    unsigned short* WaT     = (unsigned short*)alloc((size_t)96 * Dn * 2);
    unsigned short* WoutT   = (unsigned short*)alloc((size_t)Dn * Dn * 2);
    float* value  = (float*)alloc((size_t)MT * Dn * 4);
    float* offv   = (float*)alloc((size_t)MT * 192 * 4);
    float* logits = (float*)alloc((size_t)MT * 96 * 4);
    unsigned short* outp = (unsigned short*)alloc((size_t)MT * Dn * 2);

    const int n = MT * Dn;
    prep_kernel<<<(n + 255) / 256, 256, 0, stream>>>(hidden, pos, hid_bf, hs_bf, n);
    cvt_wt_kernel<<<(Dn * Dn  + 255) / 256, 256, 0, stream>>>(Wv,   WvT,   Dn, Dn);
    cvt_wt_kernel<<<(Dn * 192 + 255) / 256, 256, 0, stream>>>(Wo,   WoT,   Dn, 192);
    cvt_wt_kernel<<<(Dn * 96  + 255) / 256, 256, 0, stream>>>(Wa,   WaT,   Dn, 96);
    cvt_wt_kernel<<<(Dn * Dn  + 255) / 256, 256, 0, stream>>>(Wout, WoutT, Dn, Dn);

    dim3 blk(256);
    gemm_bf16_tdm_kernel<<<dim3(Dn  / 32, MT / 128), blk, 0, stream>>>(hid_bf, WvT, bv, value,  MT, Dn,  Dn);
    gemm_bf16_tdm_kernel<<<dim3(192 / 32, MT / 128), blk, 0, stream>>>(hs_bf,  WoT, bo, offv,   MT, 192, Dn);
    gemm_bf16_tdm_kernel<<<dim3(96  / 32, MT / 128), blk, 0, stream>>>(hs_bf,  WaT, ba, logits, MT, 96,  Dn);

    softmax_kernel<<<(MT * NHn + 255) / 256, 256, 0, stream>>>(logits, attn_out, MT * NHn);
    sample_kernel<<<(MT * NHn) / 8, 256, 0, stream>>>(value, offv, attn_out, ref, outp);

    gemm_bf16_tdm_kernel<<<dim3(Dn / 32, MT / 128), blk, 0, stream>>>(outp, WoutT, bout, out, MT, Dn, Dn);
}
